// MoEBitNetFFN_3745211482624
// MI455X (gfx1250) — 8-bit matrix operand order repaired and checked by reading, compiled, not run on hardware
//
#include <hip/hip_runtime.h>
#include <hip/hip_bf16.h>
#include <cstdint>
#include <cstddef>

#define D_MODEL 1024
#define D_FF    4096
#define NEXP    16
#define NTOK    8192
#define CAP     1536
#define NSLOT   (NEXP * CAP)

typedef int v8i __attribute__((ext_vector_type(8)));
typedef unsigned int v4u __attribute__((ext_vector_type(4)));
typedef int v4i __attribute__((ext_vector_type(4)));
typedef int v8si __attribute__((ext_vector_type(8)));

#if defined(__gfx1250__) && __has_builtin(__builtin_amdgcn_tensor_load_to_lds) && \
    __has_builtin(__builtin_amdgcn_s_wait_tensorcnt)
#define USE_TDM 1
#else
#define USE_TDM 0
#endif

__device__ __forceinline__ int q8(float v, float s) {
    float q = rintf(v * s);
    q = fminf(127.f, fmaxf(-128.f, q));
    return ((int)q) & 255;
}
__device__ __forceinline__ int pack4(float a, float b, float c, float d, float s) {
    return q8(a, s) | (q8(b, s) << 8) | (q8(c, s) << 16) | (q8(d, s) << 24);
}

#if USE_TDM
__device__ __forceinline__ void tdm_load_tile(const int8_t* gbase, int row_stride,
                                              void* lds_dst) {
    unsigned long long ga = (unsigned long long)(uintptr_t)gbase;
    unsigned lds = (unsigned)(uintptr_t)lds_dst;
    v4u g0;
    g0[0] = 1u;
    g0[1] = lds;
    g0[2] = (unsigned)ga;
    g0[3] = (unsigned)((ga >> 32) & 0x1FFFFFFu) | (2u << 30);
    v8si g1;
    g1[0] = (int)((1u << 20) | (3u << 22) | (3u << 25));
    g1[1] = (int)(64u << 16);
    g1[2] = (int)(128u << 16);
    g1[3] = (int)(64u << 16);
    g1[4] = 128;
    g1[5] = row_stride;
    g1[6] = 0;
    g1[7] = 0;
    v4i g2 = {0, 0, 0, 0};
    v4i g3 = {0, 0, 0, 0};
    v8si gx = {0, 0, 0, 0, 0, 0, 0, 0};
    __builtin_amdgcn_tensor_load_to_lds(g0, g1, g2, g3, gx, 0);
}
#endif

__global__ void init_small(int* cursor, float* psum, float* wsum) {
    int t = threadIdx.x;
    if (t < NEXP) { cursor[t] = 0; psum[t] = 0.f; }
    if (t < 2 * NEXP) wsum[t] = 0.f;
}

__global__ void fill_zero_b128(int4* p, long n4) {
    long i = (long)blockIdx.x * blockDim.x + threadIdx.x;
    long stride = (long)gridDim.x * blockDim.x;
    int4 z = make_int4(0, 0, 0, 0);
    for (; i < n4; i += stride) p[i] = z;
}

__global__ void wabs_kernel(const float* __restrict__ w, float* __restrict__ wsum) {
    __shared__ float red[256];
    const int e = blockIdx.y;
    const int t = threadIdx.x;
    const float* we = w + (long)e * D_MODEL * D_FF;
    const long base = (long)blockIdx.x * 256 + t;
    float s = 0.f;
    for (int j = 0; j < 64; ++j) s += fabsf(we[base + (long)j * 65536]);
    red[t] = s;
    __syncthreads();
    for (int s2 = 128; s2 > 0; s2 >>= 1) {
        if (t < s2) red[t] += red[t + s2];
        __syncthreads();
    }
    if (t == 0) atomicAdd(&wsum[e], red[0]);
}

__global__ void wscale_kernel(const float* __restrict__ wsum, float* __restrict__ wscale) {
    int t = threadIdx.x;
    if (t < 2 * NEXP) {
        float mean = wsum[t] / (float)((long)D_MODEL * D_FF);
        wscale[t] = fmaxf(mean, 1e-5f);
    }
}

__global__ void wquant_tr(const float* __restrict__ w, int8_t* __restrict__ wq,
                          const float* __restrict__ wscale, int sidx, int K, int N) {
    __shared__ int8_t tile[64][65];
    const int e = blockIdx.z;
    const float inv = 1.0f / wscale[sidx + e];
    const long base_in = (long)e * K * N;
    const long base_out = (long)e * N * K;
    const int k0 = blockIdx.x * 64, n0 = blockIdx.y * 64;
    const int t = threadIdx.x;
    for (int j = 0; j < 16; ++j) {
        int id = t + 256 * j;
        int k = id >> 6, n = id & 63;
        float v = w[base_in + (long)(k0 + k) * N + (n0 + n)];
        float q = rintf(v * inv);
        q = fminf(1.f, fmaxf(-1.f, q));
        tile[k][n] = (int8_t)q;
    }
    __syncthreads();
    const int n = t >> 2;
    const int kc = (t & 3) * 16;
    int word[4];
    for (int wi = 0; wi < 4; ++wi) {
        int b0 = (int)tile[kc + wi * 4 + 0][n] & 255;
        int b1 = (int)tile[kc + wi * 4 + 1][n] & 255;
        int b2 = (int)tile[kc + wi * 4 + 2][n] & 255;
        int b3 = (int)tile[kc + wi * 4 + 3][n] & 255;
        word[wi] = b0 | (b1 << 8) | (b2 << 16) | (b3 << 24);
    }
    *(int4*)(wq + base_out + (long)(n0 + n) * K + k0 + kc) =
        make_int4(word[0], word[1], word[2], word[3]);
}

__global__ void router_kernel(const float* __restrict__ x, const float* __restrict__ wr,
                              int* __restrict__ top_idx, float* __restrict__ gw,
                              float* __restrict__ sx, float* __restrict__ psum) {
    __shared__ float xs[D_MODEL];
    __shared__ float red[128];
    __shared__ float pl[128];
    __shared__ float logit[NEXP];
    __shared__ float prob[NEXP];
    const int t = threadIdx.x;
    const int tok = blockIdx.x;
    const float* xr = x + (long)tok * D_MODEL;
    float am = 0.f;
    for (int j = 0; j < 8; ++j) {
        float v = xr[t + 128 * j];
        xs[t + 128 * j] = v;
        am = fmaxf(am, fabsf(v));
    }
    red[t] = am;
    __syncthreads();
    for (int s2 = 64; s2 > 0; s2 >>= 1) {
        if (t < s2) red[t] = fmaxf(red[t], red[t + s2]);
        __syncthreads();
    }
    const int e = t >> 3, j = t & 7;
    float acc = 0.f;
    for (int k = j * 128; k < j * 128 + 128; ++k) acc += xs[k] * wr[k * NEXP + e];
    pl[t] = acc;
    __syncthreads();
    if (j == 0) {
        float s = 0.f;
        for (int q = 0; q < 8; ++q) s += pl[e * 8 + q];
        logit[e] = s;
    }
    __syncthreads();
    if (t == 0) {
        float mx = logit[0];
        for (int q = 1; q < NEXP; ++q) mx = fmaxf(mx, logit[q]);
        float den = 0.f;
        for (int q = 0; q < NEXP; ++q) { float p = __expf(logit[q] - mx); prob[q] = p; den += p; }
        float invd = 1.f / den;
        int bi = 0; float bp = -1.f;
        for (int q = 0; q < NEXP; ++q) {
            prob[q] *= invd;
            if (prob[q] > bp) { bp = prob[q]; bi = q; }
        }
        top_idx[tok] = bi;
        gw[tok] = bp / (bp + 1e-8f);
        sx[tok] = 127.0f / fmaxf(red[0], 1e-5f);
    }
    __syncthreads();
    if (t < NEXP) atomicAdd(&psum[t], prob[t]);
}

__global__ void dispatch_kernel(const float* __restrict__ x, const int* __restrict__ top_idx,
                                const float* __restrict__ sx, int* __restrict__ cursor,
                                int* __restrict__ gflat, float* __restrict__ slot_scale,
                                int8_t* __restrict__ xq) {
    __shared__ int sflat;
    __shared__ float ssc;
    const int t = threadIdx.x;
    const int tok = blockIdx.x;
    if (t == 0) {
        int e = top_idx[tok];
        int pos = atomicAdd(&cursor[e], 1);
        if (pos > CAP - 1) pos = CAP - 1;
        int flat = e * CAP + pos;
        gflat[tok] = flat;
        float s = sx[tok];
        slot_scale[flat] = s;
        ssc = s;
        sflat = flat;
    }
    __syncthreads();
    const int flat = sflat;
    const float s = ssc;
    const float* xr = x + (long)tok * D_MODEL;
    int8_t* dst = xq + (long)flat * D_MODEL;
    float4 a = ((const float4*)xr)[t * 2];
    float4 b = ((const float4*)xr)[t * 2 + 1];
    ((int2*)dst)[t] = make_int2(pack4(a.x, a.y, a.z, a.w, s), pack4(b.x, b.y, b.z, b.w, s));
}

template <int MODE>
__global__ void moe_gemm(const int8_t* __restrict__ Aq, const int8_t* __restrict__ Bq,
                         const float* __restrict__ rowscale, const float* __restrict__ wscale,
                         int sidx, const int* __restrict__ counts, int K,
                         void* __restrict__ outp) {
    const int e = blockIdx.z;
    int valid = counts[e];
    if (valid > CAP) valid = CAP;
    const int m0 = blockIdx.y * 128;
    if (m0 >= valid) return;
    const int n0 = blockIdx.x * 128;
    const int NCOLS = (MODE == 1) ? D_FF : D_MODEL;

    __shared__ __align__(16) int8_t sA[2][128 * 80];
    __shared__ __align__(16) int8_t sB[2][128 * 80];

    const int t = threadIdx.x;
    const int lane = t & 31;
    const int wv = t >> 5;
    const int wm = (wv & 3) * 32;
    const int wn = (wv >> 2) * 64;
    const int half = lane >> 4;
    const int l15 = lane & 15;

    const int8_t* gA = Aq + ((long)e * CAP + m0) * K;
    const int8_t* gB = Bq + ((long)e * NCOLS + n0) * K;

    v8i acc[2][4];
    {
        v8i z = {0, 0, 0, 0, 0, 0, 0, 0};
#pragma unroll
        for (int i = 0; i < 2; ++i)
#pragma unroll
            for (int jf = 0; jf < 4; ++jf) acc[i][jf] = z;
    }

    const int ksteps = K >> 6;

#if USE_TDM
    if (wv == 0) {
        tdm_load_tile(gA, K, &sA[0][0]);
        tdm_load_tile(gB, K, &sB[0][0]);
    }
#else
    const int ldrow = t >> 1;
    const int ldoff = (t & 1) * 32;
#endif

    for (int kt = 0; kt < ksteps; ++kt) {
        const int buf = kt & 1;
#if USE_TDM
        if (wv == 0) __builtin_amdgcn_s_wait_tensorcnt(0);
        __syncthreads();
        if (wv == 0 && kt + 1 < ksteps) {
            tdm_load_tile(gA + (kt + 1) * 64, K, &sA[buf ^ 1][0]);
            tdm_load_tile(gB + (kt + 1) * 64, K, &sB[buf ^ 1][0]);
        }
#else
        {
            const int4* pa = (const int4*)(gA + (long)ldrow * K + kt * 64 + ldoff);
            const int4* pb = (const int4*)(gB + (long)ldrow * K + kt * 64 + ldoff);
            int4 a0 = pa[0], a1 = pa[1];
            int4 b0 = pb[0], b1 = pb[1];
            *(int4*)(&sA[buf][0] + ldrow * 80 + ldoff)      = a0;
            *(int4*)(&sA[buf][0] + ldrow * 80 + ldoff + 16) = a1;
            *(int4*)(&sB[buf][0] + ldrow * 80 + ldoff)      = b0;
            *(int4*)(&sB[buf][0] + ldrow * 80 + ldoff + 16) = b1;
        }
        if (kt + 1 < ksteps) {
            __builtin_prefetch(gA + (long)ldrow * K + kt * 64 + 64 + ldoff, 0, 1);
            __builtin_prefetch(gB + (long)ldrow * K + kt * 64 + 64 + ldoff, 0, 1);
        }
        __syncthreads();
#endif
        const int8_t* bA = &sA[buf][0];
        const int8_t* bB = &sB[buf][0];

        v8i bfr[4];
#pragma unroll
        for (int fn = 0; fn < 4; ++fn) {
            const int8_t* base = bB + (wn + fn * 16 + l15) * 80 + half * 16;
            int4 d0 = *(const int4*)(base);
            int4 d1 = *(const int4*)(base + 32);
            bfr[fn][0] = d0.x; bfr[fn][1] = d0.y; bfr[fn][2] = d0.z; bfr[fn][3] = d0.w;
            bfr[fn][4] = d1.x; bfr[fn][5] = d1.y; bfr[fn][6] = d1.z; bfr[fn][7] = d1.w;
        }
        v8i afr[2];
#pragma unroll
        for (int fm = 0; fm < 2; ++fm) {
            const int8_t* base = bA + (wm + fm * 16 + l15) * 80 + half * 16;
            int4 c0 = *(const int4*)(base);
            int4 c1 = *(const int4*)(base + 32);
            afr[fm][0] = c0.x; afr[fm][1] = c0.y;
            afr[fm][2] = c0.z; afr[fm][3] = c0.w;
            afr[fm][4] = c1.x; afr[fm][5] = c1.y;
            afr[fm][6] = c1.z; afr[fm][7] = c1.w;
        }
#pragma unroll
        for (int fn = 0; fn < 4; ++fn) {
            acc[0][fn] = __builtin_amdgcn_wmma_i32_16x16x64_iu8(true, afr[0], true, bfr[fn],
                                                                acc[0][fn], false, false);
            acc[1][fn] = __builtin_amdgcn_wmma_i32_16x16x64_iu8(true, afr[1], true, bfr[fn],
                                                                acc[1][fn], false, false);
        }
#if !USE_TDM
        __syncthreads();
#endif
    }

    const float bs = wscale[sidx + e];
#pragma unroll
    for (int fm = 0; fm < 2; ++fm) {
#pragma unroll
        for (int i = 0; i < 8; ++i) {
            const long grow = (long)e * CAP + m0 + wm + fm * 16 + i + 8 * half;
            const float f = bs / rowscale[grow];
#pragma unroll
            for (int fn = 0; fn < 4; ++fn) {
                const int col = n0 + wn + fn * 16 + l15;
                float v = (float)acc[fm][fn][i] * f;
                if (MODE == 1) {
                    v = v > 0.f ? v * v : 0.f;
                    ((__hip_bfloat16*)outp)[grow * D_FF + col] = __float2bfloat16(v);
                } else {
                    ((float*)outp)[grow * D_MODEL + col] = v;
                }
            }
        }
    }
}

__global__ void hquant_kernel(const __hip_bfloat16* __restrict__ h, int8_t* __restrict__ hq,
                              float* __restrict__ hscale, const int* __restrict__ counts) {
    const long r = blockIdx.x;
    const int e = (int)(r / CAP);
    const int lp = (int)(r % CAP);
    int valid = counts[e];
    if (valid > CAP) valid = CAP;
    if (lp >= valid) return;
    __shared__ float red[256];
    const int t = threadIdx.x;
    const __hip_bfloat16* row = h + r * D_FF;
    float v[16];
    float am = 0.f;
    for (int j = 0; j < 16; ++j) {
        v[j] = __bfloat162float(row[t + 256 * j]);
        am = fmaxf(am, fabsf(v[j]));
    }
    red[t] = am;
    __syncthreads();
    for (int s2 = 128; s2 > 0; s2 >>= 1) {
        if (t < s2) red[t] = fmaxf(red[t], red[t + s2]);
        __syncthreads();
    }
    const float s = 127.f / fmaxf(red[0], 1e-5f);
    int8_t* orow = hq + r * D_FF;
    for (int j = 0; j < 16; ++j) {
        float q = rintf(v[j] * s);
        q = fminf(127.f, fmaxf(-128.f, q));
        orow[t + 256 * j] = (int8_t)q;
    }
    if (t == 0) hscale[r] = s;
}

__global__ void gather_kernel(const float* __restrict__ eo, const int* __restrict__ gflat,
                              const float* __restrict__ gw, float* __restrict__ out) {
    const int tok = blockIdx.x;
    const int t = threadIdx.x;
    const long flat = gflat[tok];
    const float wgt = gw[tok];
    float4 a = ((const float4*)(eo + flat * D_MODEL))[t];
    a.x *= wgt; a.y *= wgt; a.z *= wgt; a.w *= wgt;
    ((float4*)(out + (long)tok * D_MODEL))[t] = a;
}

__global__ void aux_kernel(const int* __restrict__ counts, const float* __restrict__ psum,
                           float* __restrict__ out_aux) {
    if (threadIdx.x == 0) {
        const float invN = 1.0f / (float)NTOK;
        float a = 0.f;
        for (int e = 0; e < NEXP; ++e) a += ((float)counts[e] * invN) * (psum[e] * invN);
        out_aux[0] = (float)NEXP * a;
    }
}

extern "C" void kernel_launch(void* const* d_in, const int* in_sizes, int n_in,
                              void* d_out, int out_size, void* d_ws, size_t ws_size,
                              hipStream_t stream) {
    (void)in_sizes; (void)n_in; (void)out_size; (void)ws_size;
    const float* x  = (const float*)d_in[0];
    const float* wr = (const float*)d_in[1];
    const float* w1 = (const float*)d_in[2];
    const float* w2 = (const float*)d_in[3];
    float* out = (float*)d_out;

    char* ws = (char*)d_ws;
    size_t off = 0;
    auto carve = [&](size_t bytes) -> void* {
        off = (off + 255) & ~(size_t)255;
        void* p = ws + off;
        off += bytes;
        return p;
    };
    float* wsum       = (float*)carve(32 * sizeof(float));
    float* wscale     = (float*)carve(32 * sizeof(float));
    int*   cursor     = (int*)carve(NEXP * sizeof(int));
    float* psum       = (float*)carve(NEXP * sizeof(float));
    int*   topidx     = (int*)carve((size_t)NTOK * sizeof(int));
    float* gw         = (float*)carve((size_t)NTOK * sizeof(float));
    float* sx         = (float*)carve((size_t)NTOK * sizeof(float));
    int*   gflat      = (int*)carve((size_t)NTOK * sizeof(int));
    float* slot_scale = (float*)carve((size_t)NSLOT * sizeof(float));
    float* hscale     = (float*)carve((size_t)NSLOT * sizeof(float));
    int8_t* w1q       = (int8_t*)carve((size_t)NEXP * D_MODEL * D_FF);
    int8_t* w2q       = (int8_t*)carve((size_t)NEXP * D_MODEL * D_FF);
    int8_t* xq        = (int8_t*)carve((size_t)NSLOT * D_MODEL);
    int8_t* hq        = (int8_t*)carve((size_t)NSLOT * D_FF);
    __hip_bfloat16* h = (__hip_bfloat16*)carve((size_t)NSLOT * D_FF * 2);
    float* eo         = (float*)carve((size_t)NSLOT * D_MODEL * sizeof(float));

    init_small<<<1, 64, 0, stream>>>(cursor, psum, wsum);
    fill_zero_b128<<<4096, 256, 0, stream>>>((int4*)xq, (long)NSLOT * D_MODEL / 16);

    wabs_kernel<<<dim3(256, NEXP), 256, 0, stream>>>(w1, wsum);
    wabs_kernel<<<dim3(256, NEXP), 256, 0, stream>>>(w2, wsum + NEXP);
    wscale_kernel<<<1, 32, 0, stream>>>(wsum, wscale);

    wquant_tr<<<dim3(D_MODEL / 64, D_FF / 64, NEXP), 256, 0, stream>>>(w1, w1q, wscale, 0,
                                                                       D_MODEL, D_FF);
    wquant_tr<<<dim3(D_FF / 64, D_MODEL / 64, NEXP), 256, 0, stream>>>(w2, w2q, wscale, NEXP,
                                                                       D_FF, D_MODEL);

    router_kernel<<<NTOK, 128, 0, stream>>>(x, wr, topidx, gw, sx, psum);
    dispatch_kernel<<<NTOK, 128, 0, stream>>>(x, topidx, sx, cursor, gflat, slot_scale, xq);

    moe_gemm<1><<<dim3(D_FF / 128, CAP / 128, NEXP), 256, 0, stream>>>(
        xq, w1q, slot_scale, wscale, 0, cursor, D_MODEL, h);

    hquant_kernel<<<NSLOT, 256, 0, stream>>>(h, hq, hscale, cursor);

    moe_gemm<2><<<dim3(D_MODEL / 128, CAP / 128, NEXP), 256, 0, stream>>>(
        hq, w2q, hscale, wscale, NEXP, cursor, D_FF, eo);

    gather_kernel<<<NTOK, 256, 0, stream>>>(eo, gflat, gw, out);
    aux_kernel<<<1, 32, 0, stream>>>(cursor, psum, out + (size_t)NTOK * D_MODEL);
}
